// G2GDecoder_5798205849964
// MI455X (gfx1250) — matrix operand order repaired and checked by reading, not compiled, not run on hardware
//
#include <hip/hip_runtime.h>
#include <hip/hip_bf16.h>
#include <math.h>

// ---- sizes (match reference) ----
#define NLG   65536
#define ELG   131072
#define TS    8
#define MM    4096
#define DF    256
#define DMSG  256
#define DH    512
#define DXT   256
#define DXG   256
#define DZD   512
#define DZL   512
#define NLAB  512
#define NT    2048
#define NG    4096

typedef __bf16 bf16_t;
typedef __attribute__((ext_vector_type(16))) __bf16 v16bf;
typedef __attribute__((ext_vector_type(8)))  float  v8f;

union Frag { v16bf v; unsigned u[8]; };

// A fragment: 16x32 (MxK) bf16, row-major with leading dim lda.
// Lane l: row given; k-half = (l>>4)*8; VGPR i holds K pair (i&3)*2 + (i>>2)*16.
__device__ __forceinline__ v16bf load_a_frag(const bf16_t* A, int lda, int row, int kbase, int lane) {
  Frag f;
  const bf16_t* p = A + (size_t)row * lda + kbase + ((lane >> 4) << 3);
#pragma unroll
  for (int i = 0; i < 8; ++i) {
    int k = ((i & 3) << 1) + ((i >> 2) << 4);
    f.u[i] = *(const unsigned*)(p + k);
  }
  return f.v;
}

// B fragment: 32x16 (KxN). Weights pre-packed N-major: element(k,n) = W[n*ldw + k], ldw == K.
// Lane l: n = nbase + (l&15); k-half = (l>>4)*8; VGPR i holds K pair (i&3)*2 + (i>>2)*16.
__device__ __forceinline__ v16bf load_b_frag(const bf16_t* W, int ldw, int kbase, int nbase, int lane) {
  Frag f;
  const bf16_t* p = W + (size_t)(nbase + (lane & 15)) * ldw + kbase + ((lane >> 4) << 3);
#pragma unroll
  for (int i = 0; i < 8; ++i) f.u[i] = *(const unsigned*)(p + ((i & 3) << 1) + ((i >> 2) << 4));
  return f.v;
}

#define WMMA_BF16(A, B, C) \
  __builtin_amdgcn_wmma_f32_16x16x32_bf16(false, (A), false, (B), (short)0, (C), false, false)

__device__ __forceinline__ float act_apply(float x, int act) {
  if (act == 1) return 1.0f / (1.0f + __expf(-x));
  if (act == 2) return fmaxf(x, 0.0f);
  if (act == 3) return tanhf(x);
  return x;
}

__device__ __forceinline__ float logsig(float x) {
  return (x < 0.0f) ? (x - log1pf(expf(x))) : (-log1pf(expf(-x)));
}

// ---------------- generic bf16-WMMA GEMM ----------------
// C(M,N) = act( A1(M,K1)@W1p + [A2(M,K2)@W2p] + bias )
// W*p packed N-major (N x K). Block tile 64x128 (8 waves, each 32x32 via 2x2 frags).
// grid = (N/128, M/64), 256 threads.
__global__ __launch_bounds__(256) void k_gemm(
    const bf16_t* __restrict__ A1, int K1, const bf16_t* __restrict__ W1,
    const bf16_t* __restrict__ A2, int K2, const bf16_t* __restrict__ W2,
    const float* __restrict__ bias, int biasMode, int act,
    float* __restrict__ outF, bf16_t* __restrict__ outB, int ldc)
{
  const int lane   = threadIdx.x & 31;
  const int wave   = threadIdx.x >> 5;
  const int tile_m = blockIdx.y * 64 + (wave & 1) * 32;
  const int tile_n = blockIdx.x * 128 + (wave >> 1) * 32;
  const int row0   = tile_m + (lane & 15);
  const int row1   = row0 + 16;

  v8f acc00 = {}, acc01 = {}, acc10 = {}, acc11 = {};
  for (int k = 0; k < K1; k += 32) {
    v16bf a0 = load_a_frag(A1, K1, row0, k, lane);
    v16bf a1 = load_a_frag(A1, K1, row1, k, lane);
    v16bf b0 = load_b_frag(W1, K1, k, tile_n, lane);
    v16bf b1 = load_b_frag(W1, K1, k, tile_n + 16, lane);
    acc00 = WMMA_BF16(a0, b0, acc00);
    acc01 = WMMA_BF16(a0, b1, acc01);
    acc10 = WMMA_BF16(a1, b0, acc10);
    acc11 = WMMA_BF16(a1, b1, acc11);
  }
  if (A2) {
    for (int k = 0; k < K2; k += 32) {
      v16bf a0 = load_a_frag(A2, K2, row0, k, lane);
      v16bf a1 = load_a_frag(A2, K2, row1, k, lane);
      v16bf b0 = load_b_frag(W2, K2, k, tile_n, lane);
      v16bf b1 = load_b_frag(W2, K2, k, tile_n + 16, lane);
      acc00 = WMMA_BF16(a0, b0, acc00);
      acc01 = WMMA_BF16(a0, b1, acc01);
      acc10 = WMMA_BF16(a1, b0, acc10);
      acc11 = WMMA_BF16(a1, b1, acc11);
    }
  }
  const int nloc = lane & 15;
  const int mloc = (lane >> 4) << 3;
  v8f accs[2][2] = {{acc00, acc01}, {acc10, acc11}};
#pragma unroll
  for (int j = 0; j < 2; ++j) {
    const int n = tile_n + j * 16 + nloc;
    const float bv = (biasMode == 1) ? bias[n] : (biasMode == 2 ? bias[0] : 0.0f);
#pragma unroll
    for (int i = 0; i < 2; ++i) {
#pragma unroll
      for (int r = 0; r < 8; ++r) {
        int m = tile_m + i * 16 + r + mloc;
        float v = act_apply(accs[i][j][r] + bv, act);
        if (outF) outF[(size_t)m * ldc + n] = v;
        if (outB) outB[(size_t)m * ldc + n] = (bf16_t)v;
      }
    }
  }
}

// ---- fused per-edge reset gate: r = sigmoid(Fd[src]@Wr + H[src]@Ur + br);
//      srh_full[dst] += r * Hf[src].  grid = (256/128, ELG/64). Wr/Ur packed N-major. ----
__global__ __launch_bounds__(256) void k_edge_gate(
    const bf16_t* __restrict__ Fd, const bf16_t* __restrict__ Hb, const float* __restrict__ Hf,
    const bf16_t* __restrict__ Wr, const bf16_t* __restrict__ Ur, const float* __restrict__ br,
    const int* __restrict__ src, const int* __restrict__ dst, float* __restrict__ outAcc)
{
  const int lane   = threadIdx.x & 31;
  const int wave   = threadIdx.x >> 5;
  const int tile_m = blockIdx.y * 64 + (wave & 1) * 32;
  const int tile_n = blockIdx.x * 128 + (wave >> 1) * 32;
  const int arow0  = src[tile_m + (lane & 15)];
  const int arow1  = src[tile_m + 16 + (lane & 15)];

  v8f acc00 = {}, acc01 = {}, acc10 = {}, acc11 = {};
  for (int k = 0; k < DMSG; k += 32) {
    v16bf a0 = load_a_frag(Fd, DMSG, arow0, k, lane);
    v16bf a1 = load_a_frag(Fd, DMSG, arow1, k, lane);
    v16bf b0 = load_b_frag(Wr, DMSG, k, tile_n, lane);
    v16bf b1 = load_b_frag(Wr, DMSG, k, tile_n + 16, lane);
    acc00 = WMMA_BF16(a0, b0, acc00);
    acc01 = WMMA_BF16(a0, b1, acc01);
    acc10 = WMMA_BF16(a1, b0, acc10);
    acc11 = WMMA_BF16(a1, b1, acc11);
    a0 = load_a_frag(Hb, DMSG, arow0, k, lane);
    a1 = load_a_frag(Hb, DMSG, arow1, k, lane);
    b0 = load_b_frag(Ur, DMSG, k, tile_n, lane);
    b1 = load_b_frag(Ur, DMSG, k, tile_n + 16, lane);
    acc00 = WMMA_BF16(a0, b0, acc00);
    acc01 = WMMA_BF16(a0, b1, acc01);
    acc10 = WMMA_BF16(a1, b0, acc10);
    acc11 = WMMA_BF16(a1, b1, acc11);
  }
  const int nloc = lane & 15;
  const int mloc = (lane >> 4) << 3;
  v8f accs[2][2] = {{acc00, acc01}, {acc10, acc11}};
#pragma unroll
  for (int j = 0; j < 2; ++j) {
    const int n = tile_n + j * 16 + nloc;
    const float bv = br[n];
#pragma unroll
    for (int i = 0; i < 2; ++i) {
#pragma unroll
      for (int r = 0; r < 8; ++r) {
        int m  = tile_m + i * 16 + r + mloc;
        int sm = src[m];
        int dm = dst[m];
        float g = 1.0f / (1.0f + __expf(-(accs[i][j][r] + bv)));
        atomicAdd(&outAcc[(size_t)dm * DMSG + n], g * Hf[(size_t)sm * DMSG + n]);
      }
    }
  }
}

// ---------------- elementwise helpers ----------------
__global__ void k_zero(float* p, long n) {
  long i = (long)blockIdx.x * blockDim.x + threadIdx.x;
  if (i < n) p[i] = 0.0f;
}
__global__ void k_zero16(bf16_t* p, long n) {
  long i = (long)blockIdx.x * blockDim.x + threadIdx.x;
  if (i < n) p[i] = (bf16_t)0.0f;
}
__global__ void k_cvt(const float* __restrict__ s, bf16_t* __restrict__ d, long n) {
  long i = (long)blockIdx.x * blockDim.x + threadIdx.x;
  if (i < n) d[i] = (bf16_t)s[i];
}
// pack W (K x N, f32 row-major) -> Wt (N x K, bf16): Wt[n*K+k] = W[k*N+n]
__global__ void k_packT(const float* __restrict__ W, bf16_t* __restrict__ Wt, int K, int N) {
  long i = (long)blockIdx.x * blockDim.x + threadIdx.x;
  if (i >= (long)K * N) return;
  int n = (int)(i / K), k = (int)(i % K);
  Wt[i] = (bf16_t)W[(size_t)k * N + n];
}
// scatter-add: out[dst[e]] += h[src[e]] over all (edge, col)
__global__ void k_scatter(const float* __restrict__ h, const int* __restrict__ src,
                          const int* __restrict__ dst, float* __restrict__ out, long total) {
  long i = (long)blockIdx.x * blockDim.x + threadIdx.x;
  if (i >= total) return;
  int e = (int)(i >> 8), c = (int)(i & 255);
  atomicAdd(&out[(size_t)dst[e] * DMSG + c], h[(size_t)src[e] * DMSG + c]);
}
__global__ void k_gather(const float* __restrict__ s, const int* __restrict__ idx,
                         float* __restrict__ oF, bf16_t* __restrict__ oB, long total) {
  long i = (long)blockIdx.x * blockDim.x + threadIdx.x;
  if (i >= total) return;
  int r = (int)(i >> 8), c = (int)(i & 255);
  float v = s[(size_t)idx[r] * DMSG + c];
  if (oF) oF[i] = v;
  if (oB) oB[i] = (bf16_t)v;
}
// h[e] = (1-z)*s + z*ht  (f32 master + bf16 mirror)
__global__ void k_update(float* __restrict__ h, bf16_t* __restrict__ hb, const int* __restrict__ e,
                         const float* __restrict__ z, const float* __restrict__ s,
                         const float* __restrict__ ht, long total) {
  long i = (long)blockIdx.x * blockDim.x + threadIdx.x;
  if (i >= total) return;
  int r = (int)(i >> 8), c = (int)(i & 255);
  int row = e[r];
  float zz = z[i];
  float v  = (1.0f - zz) * s[i] + zz * ht[i];
  h[(size_t)row * DMSG + c]  = v;
  hb[(size_t)row * DMSG + c] = (bf16_t)v;
}
// per-row softmax over L cols; write bf16 attention weights
__global__ __launch_bounds__(256) void k_softmax(const float* __restrict__ sc, bf16_t* __restrict__ att, int L) {
  const int row = blockIdx.x, tid = threadIdx.x;
  const float* p = sc + (size_t)row * L;
  __shared__ float red[256];
  float mx = -INFINITY;
  for (int i = tid; i < L; i += 256) mx = fmaxf(mx, p[i]);
  red[tid] = mx; __syncthreads();
  for (int s = 128; s > 0; s >>= 1) { if (tid < s) red[tid] = fmaxf(red[tid], red[tid + s]); __syncthreads(); }
  mx = red[0]; __syncthreads();
  float sum = 0.0f;
  for (int i = tid; i < L; i += 256) sum += __expf(p[i] - mx);
  red[tid] = sum; __syncthreads();
  for (int s = 128; s > 0; s >>= 1) { if (tid < s) red[tid] += red[tid + s]; __syncthreads(); }
  float inv = 1.0f / red[0];
  for (int i = tid; i < L; i += 256) att[(size_t)row * L + i] = (bf16_t)(__expf(p[i] - mx) * inv);
}
// topo: pr = z_t[row].u_d + b_d3 ; loss0 += -(lab*logsig(pr)+(1-lab)*logsig(1-pr))
__global__ __launch_bounds__(256) void k_topo(const float* __restrict__ zt, const float* __restrict__ ud,
                                              const float* __restrict__ bd3, const float* __restrict__ lab,
                                              float* __restrict__ loss) {
  const int row = blockIdx.x, tid = threadIdx.x;
  __shared__ float red[256];
  const float* p = zt + (size_t)row * DZD;
  red[tid] = p[tid] * ud[tid] + p[tid + 256] * ud[tid + 256];
  __syncthreads();
  for (int s = 128; s > 0; s >>= 1) { if (tid < s) red[tid] += red[tid + s]; __syncthreads(); }
  if (tid == 0) {
    float pr = red[0] + bd3[0];
    float l  = lab[row];
    atomicAdd(loss, -(l * logsig(pr) + (1.0f - l) * logsig(1.0f - pr)));
  }
}
// ce: loss1 += logsumexp(q[row]) - mean(q[row])
__global__ __launch_bounds__(256) void k_ce(const float* __restrict__ q, float* __restrict__ loss) {
  const int row = blockIdx.x, tid = threadIdx.x;
  const float* p = q + (size_t)row * NLAB;
  __shared__ float red[256];
  float mx = fmaxf(p[tid], p[tid + 256]);
  red[tid] = mx; __syncthreads();
  for (int s = 128; s > 0; s >>= 1) { if (tid < s) red[tid] = fmaxf(red[tid], red[tid + s]); __syncthreads(); }
  mx = red[0]; __syncthreads();
  float sum = __expf(p[tid] - mx) + __expf(p[tid + 256] - mx);
  float tot = p[tid] + p[tid + 256];
  red[tid] = sum; __syncthreads();
  for (int s = 128; s > 0; s >>= 1) { if (tid < s) red[tid] += red[tid + s]; __syncthreads(); }
  float sumall = red[0]; __syncthreads();
  red[tid] = tot; __syncthreads();
  for (int s = 128; s > 0; s >>= 1) { if (tid < s) red[tid] += red[tid + s]; __syncthreads(); }
  if (tid == 0) atomicAdd(loss, mx + __logf(sumall) - red[0] / (float)NLAB);
}

extern "C" void kernel_launch(void* const* d_in, const int* in_sizes, int n_in,
                              void* d_out, int out_size, void* d_ws, size_t ws_size,
                              hipStream_t stream) {
  (void)in_sizes; (void)n_in; (void)out_size; (void)ws_size;
  const float* f_src = (const float*)d_in[0];
  const float* f_dst = (const float*)d_in[1];
  const float* x_T   = (const float*)d_in[2];
  const float* x_G   = (const float*)d_in[3];
  const float *wz = (const float*)d_in[4],  *uz = (const float*)d_in[5],  *bz = (const float*)d_in[6];
  const float *wr = (const float*)d_in[7],  *ur = (const float*)d_in[8],  *br = (const float*)d_in[9];
  const float *w  = (const float*)d_in[10], *u  = (const float*)d_in[11], *b  = (const float*)d_in[12];
  const float *w_d1 = (const float*)d_in[13], *w_d2 = (const float*)d_in[14], *b_d1 = (const float*)d_in[15];
  const float *a_dT = (const float*)d_in[16], *a_dG = (const float*)d_in[17];
  const float *w_d3 = (const float*)d_in[18], *w_d4 = (const float*)d_in[19], *b_d2 = (const float*)d_in[20];
  const float *u_d  = (const float*)d_in[21], *b_d3 = (const float*)d_in[22];
  const float *w_l1 = (const float*)d_in[23], *w_l2 = (const float*)d_in[24], *b_l1 = (const float*)d_in[25];
  const float *a_lT = (const float*)d_in[26], *a_lG = (const float*)d_in[27];
  const float *u_l  = (const float*)d_in[28], *b_l2 = (const float*)d_in[29];
  const int*   lg_src = (const int*)d_in[30];
  const int*   lg_dst = (const int*)d_in[31];
  const int*   eids   = (const int*)d_in[32];
  const float* expand = (const float*)d_in[33];

  float* h    = (float*)d_out;                 // (NLG, DMSG) master h, becomes h_fin
  float* loss = h + (size_t)NLG * DMSG;        // 2 floats

  // ---- carve workspace ----
  char* base = (char*)d_ws; size_t off = 0;
  auto carve = [&](size_t bytes) -> void* {
    void* p = base + off; off += (bytes + 255) & ~(size_t)255; return p;
  };
  // packed (N x K) bf16 weights
  bf16_t* wz_p  = (bf16_t*)carve(DF * DMSG * 2);
  bf16_t* uz_p  = (bf16_t*)carve(DMSG * DMSG * 2);
  bf16_t* wr_p  = (bf16_t*)carve(DF * DMSG * 2);
  bf16_t* ur_p  = (bf16_t*)carve(DMSG * DMSG * 2);
  bf16_t* w_p   = (bf16_t*)carve(DF * DMSG * 2);
  bf16_t* u_p   = (bf16_t*)carve(DMSG * DMSG * 2);
  bf16_t* wd1_p = (bf16_t*)carve(DF * DH * 2);
  bf16_t* wd2_p = (bf16_t*)carve(DMSG * DH * 2);
  bf16_t* adT_p = (bf16_t*)carve(DH * DXT * 2);
  bf16_t* adG_p = (bf16_t*)carve(DH * DXG * 2);
  bf16_t* wd3_p = (bf16_t*)carve(DH * DZD * 2);
  bf16_t* wd4_p = (bf16_t*)carve(512 * DZD * 2);
  bf16_t* wl1_p = (bf16_t*)carve(DMSG * DZL * 2);
  bf16_t* wl2_p = (bf16_t*)carve(512 * DZL * 2);
  bf16_t* alT_p = (bf16_t*)carve(DMSG * DXT * 2);
  bf16_t* alG_p = (bf16_t*)carve(DMSG * DXG * 2);
  bf16_t* ul_p  = (bf16_t*)carve(DZL * NLAB * 2);
  // x in both layouts
  bf16_t* xT_b  = (bf16_t*)carve((size_t)NT * DXT * 2);  // (cand x 256): B for scores
  bf16_t* xG_b  = (bf16_t*)carve((size_t)NG * DXG * 2);
  bf16_t* xTt_p = (bf16_t*)carve((size_t)DXT * NT * 2);  // (256 x cand): B for context
  bf16_t* xGt_p = (bf16_t*)carve((size_t)DXG * NG * 2);
  // A-side data
  bf16_t* fdst_b= (bf16_t*)carve((size_t)NLG * DF * 2);
  bf16_t* h_b   = (bf16_t*)carve((size_t)NLG * DMSG * 2);
  float*  sfull = (float*) carve((size_t)NLG * DMSG * 4);
  float*  scores= (float*) carve((size_t)MM * NG * 4);   // also reused as q
  bf16_t* att_b = (bf16_t*)carve((size_t)MM * NG * 2);
  bf16_t* fs_b  = (bf16_t*)carve((size_t)MM * DF * 2);
  float*  sm_f  = (float*) carve((size_t)MM * DMSG * 4);
  bf16_t* sm_b  = (bf16_t*)carve((size_t)MM * DMSG * 2);
  bf16_t* srh_b = (bf16_t*)carve((size_t)MM * DMSG * 2);
  float*  z_f   = (float*) carve((size_t)MM * DMSG * 4);
  float*  ht_f  = (float*) carve((size_t)MM * DMSG * 4);
  bf16_t* sumh_b= (bf16_t*)carve((size_t)MM * DMSG * 2);
  bf16_t* hd_b  = (bf16_t*)carve((size_t)MM * DH * 2);
  bf16_t* hA_b  = (bf16_t*)carve((size_t)MM * 256 * 2);
  bf16_t* ct_b  = (bf16_t*)carve((size_t)MM * 512 * 2);
  float*  zt_f  = (float*) carve((size_t)MM * DZD * 4);
  bf16_t* msg_b = (bf16_t*)carve((size_t)MM * DMSG * 2);
  bf16_t* cl_b  = (bf16_t*)carve((size_t)MM * 512 * 2);
  bf16_t* zl_b  = (bf16_t*)carve((size_t)MM * DZL * 2);
  float*  q_f   = scores;                                 // alias (disjoint in time)

  auto CVT = [&](const float* s, bf16_t* d, long n) {
    k_cvt<<<(unsigned)((n + 255) / 256), 256, 0, stream>>>(s, d, n);
  };
  auto PACK = [&](const float* W, bf16_t* Wt, int K, int N) {
    long n = (long)K * N;
    k_packT<<<(unsigned)((n + 255) / 256), 256, 0, stream>>>(W, Wt, K, N);
  };
  auto ZERO = [&](float* p, long n) {
    k_zero<<<(unsigned)((n + 255) / 256), 256, 0, stream>>>(p, n);
  };
  auto GEMM = [&](const bf16_t* A1, int K1, const bf16_t* W1,
                  const bf16_t* A2, int K2, const bf16_t* W2,
                  const float* bias, int bm, int act, int Mrows, int Ncols,
                  float* oF, bf16_t* oB, int ldc) {
    dim3 g(Ncols / 128, Mrows / 64);
    k_gemm<<<g, 256, 0, stream>>>(A1, K1, W1, A2, K2, W2, bias, bm, act, oF, oB, ldc);
  };
  // attention block: ctxOut(:,0:256 of 512) = softmax(Q@Aw @ X^T) @ X
  auto ATT = [&](const bf16_t* Q, int Kq, const bf16_t* Awp, const bf16_t* Xb, const bf16_t* Xtp,
                 int Ncand, bf16_t* ctxOut) {
    GEMM(Q, Kq, Awp, nullptr, 0, nullptr, nullptr, 0, 0, MM, 256, nullptr, hA_b, 256);
    GEMM(hA_b, 256, Xb, nullptr, 0, nullptr, nullptr, 0, 0, MM, Ncand, scores, nullptr, Ncand);
    k_softmax<<<MM, 256, 0, stream>>>(scores, att_b, Ncand);
    GEMM(att_b, Ncand, Xtp, nullptr, 0, nullptr, nullptr, 0, 0, MM, 256, nullptr, ctxOut, 512);
  };

  // ---- one-time (per launch) packs & init ----
  PACK(wz, wz_p, DF, DMSG);     PACK(uz, uz_p, DMSG, DMSG);
  PACK(wr, wr_p, DF, DMSG);     PACK(ur, ur_p, DMSG, DMSG);
  PACK(w,  w_p,  DF, DMSG);     PACK(u,  u_p,  DMSG, DMSG);
  PACK(w_d1, wd1_p, DF, DH);    PACK(w_d2, wd2_p, DMSG, DH);
  PACK(a_dT, adT_p, DH, DXT);   PACK(a_dG, adG_p, DH, DXG);
  PACK(w_d3, wd3_p, DH, DZD);   PACK(w_d4, wd4_p, 512, DZD);
  PACK(w_l1, wl1_p, DMSG, DZL); PACK(w_l2, wl2_p, 512, DZL);
  PACK(a_lT, alT_p, DMSG, DXT); PACK(a_lG, alG_p, DMSG, DXG);
  PACK(u_l, ul_p, DZL, NLAB);
  CVT(x_T, xT_b, (long)NT * DXT); CVT(x_G, xG_b, (long)NG * DXG);
  PACK(x_T, xTt_p, NT, DXT);      PACK(x_G, xGt_p, NG, DXG);
  CVT(f_dst, fdst_b, (long)NLG * DF);
  ZERO(h, (long)NLG * DMSG + 2);
  k_zero16<<<(unsigned)(((long)NLG * DMSG + 255) / 256), 256, 0, stream>>>(h_b, (long)NLG * DMSG);

  const long mTot = (long)MM * DMSG;
  const unsigned mBlk = (unsigned)((mTot + 255) / 256);
  const long eTot = (long)ELG * DMSG;

  for (int t = 0; t < TS; ++t) {
    const int*   e   = eids   + (size_t)t * MM;
    const float* lab = expand + (size_t)t * MM;

    // s = segsum(h[lg_src] -> lg_dst)[e]
    ZERO(sfull, (long)NLG * DMSG);
    k_scatter<<<(unsigned)(eTot / 256), 256, 0, stream>>>(h, lg_src, lg_dst, sfull, eTot);
    k_gather<<<mBlk, 256, 0, stream>>>(sfull, e, sm_f, sm_b, mTot);
    k_gather<<<mBlk, 256, 0, stream>>>(f_src, e, nullptr, fs_b, mTot);

    // z = sigmoid(fs@wz + s@uz + bz)
    GEMM(fs_b, DF, wz_p, sm_b, DMSG, uz_p, bz, 1, 1, MM, DMSG, z_f, nullptr, DMSG);

    // srh = segsum(sigmoid(fd[src]@wr + h[src]@ur + br) * h[src] -> dst)[e]
    ZERO(sfull, (long)NLG * DMSG);
    k_edge_gate<<<dim3(DMSG / 128, ELG / 64), 256, 0, stream>>>(fdst_b, h_b, h, wr_p, ur_p, br,
                                                                lg_src, lg_dst, sfull);
    k_gather<<<mBlk, 256, 0, stream>>>(sfull, e, nullptr, srh_b, mTot);

    // h_tilde = tanh(fs@w + srh@u + b); h[e] = (1-z)*s + z*h_tilde
    GEMM(fs_b, DF, w_p, srh_b, DMSG, u_p, b, 1, 3, MM, DMSG, ht_f, nullptr, DMSG);
    k_update<<<mBlk, 256, 0, stream>>>(h, h_b, e, z_f, sm_f, ht_f, mTot);

    // sum_h = segsum(new h)[e]
    ZERO(sfull, (long)NLG * DMSG);
    k_scatter<<<(unsigned)(eTot / 256), 256, 0, stream>>>(h, lg_src, lg_dst, sfull, eTot);
    k_gather<<<mBlk, 256, 0, stream>>>(sfull, e, nullptr, sumh_b, mTot);

    // topology head
    GEMM(fs_b, DF, wd1_p, sumh_b, DMSG, wd2_p, b_d1, 1, 2, MM, DH, nullptr, hd_b, DH);
    ATT(hd_b, DH, adT_p, xT_b, xTt_p, NT, ct_b);
    ATT(hd_b, DH, adG_p, xG_b, xGt_p, NG, ct_b + 256);
    GEMM(hd_b, DH, wd3_p, ct_b, 512, wd4_p, b_d2, 1, 2, MM, DZD, zt_f, nullptr, DZD);
    k_topo<<<MM, 256, 0, stream>>>(zt_f, u_d, b_d3, lab, loss + 0);

    // label head
    k_gather<<<mBlk, 256, 0, stream>>>(h, e, nullptr, msg_b, mTot);
    ATT(msg_b, DMSG, alT_p, xT_b, xTt_p, NT, cl_b);
    ATT(msg_b, DMSG, alG_p, xG_b, xGt_p, NG, cl_b + 256);
    GEMM(msg_b, DMSG, wl1_p, cl_b, 512, wl2_p, b_l1, 1, 2, MM, DZL, nullptr, zl_b, DZL);
    GEMM(zl_b, DZL, ul_p, nullptr, 0, nullptr, b_l2, 2, 0, MM, NLAB, q_f, nullptr, NLAB);
    k_ce<<<MM, 256, 0, stream>>>(q_f, loss + 1);
  }
}
